// Bidirectional_LSTM_Decoder_21234318311670
// MI455X (gfx1250) — matrix operand order repaired and checked by reading, not compiled, not run on hardware
//
#include <hip/hip_runtime.h>

#define Hh 512
#define Ll 2
#define Bb 32
#define Ss 512
#define Dd 512
#define G4 2048                 // 4*H
#define KT 32                   // K tiles of 32 (K = 1024 = [x|h] concat)
#define FRAG (128*32*32*16)     // halves per packed weight matrix = 2^21
#define QSTRIDE (32*KT*32*16)   // halves between gate quadrants = 524288 (1MB)

typedef _Float16 v16h __attribute__((ext_vector_type(16)));
typedef _Float16 h8   __attribute__((ext_vector_type(8)));
typedef float    v8f  __attribute__((ext_vector_type(8)));

__device__ __forceinline__ float sigf(float x) { return 1.f / (1.f + __expf(-x)); }
__device__ __forceinline__ float tanh_(float x) { float e = __expf(2.f * x); return 1.f - 2.f / (e + 1.f); }

// ---------------- prep kernels ----------------

__global__ void cvt_x(const float* __restrict__ x, _Float16* __restrict__ xh, int n) {
    for (int i = blockIdx.x * blockDim.x + threadIdx.x; i < n; i += gridDim.x * blockDim.x)
        xh[i] = (_Float16)x[i];
}

// Pack [Wih | Whh] (4H x 1024) into WMMA B-fragment order per (layer,dir):
// index = ((nTile*KT + kt)*32 + lane)*16 + half ; lane<16 -> K 0..15, lane>=16 -> K 16..31 of the kTile.
__global__ void pack_w(const float* __restrict__ Wih_f, const float* __restrict__ Whh_f,
                       const float* __restrict__ Wih_b, const float* __restrict__ Whh_b,
                       _Float16* __restrict__ wp) {
    int idx  = blockIdx.x * blockDim.x + threadIdx.x;   // < 4*FRAG
    int mat  = idx >> 21;
    int rem  = idx & (FRAG - 1);
    int h    = rem & 15;
    int lane = (rem >> 4) & 31;
    int kt   = (rem >> 9) & 31;
    int nt   = rem >> 14;
    int layer = mat >> 1, dir = mat & 1;
    int n = nt * 16 + (lane & 15);
    int k = kt * 32 + ((lane >> 4) ? 16 : 0) + h;
    const float* Wih = dir ? Wih_b : Wih_f;
    const float* Whh = dir ? Whh_b : Whh_f;
    float v = (k < 512) ? Wih[(size_t)layer * G4 * 512 + (size_t)n * 512 + k]
                        : Whh[(size_t)layer * G4 * 512 + (size_t)n * 512 + (k - 512)];
    wp[idx] = (_Float16)v;
}

__global__ void pack_bias(const float* __restrict__ bih_f, const float* __restrict__ bhh_f,
                          const float* __restrict__ bih_b, const float* __restrict__ bhh_b,
                          float* __restrict__ bias) {
    int idx = blockIdx.x * blockDim.x + threadIdx.x;    // < 2*2*2048
    int n = idx & 2047;
    int dir = (idx >> 11) & 1;
    int layer = idx >> 12;
    const float* bi = dir ? bih_b : bih_f;
    const float* bh = dir ? bhh_b : bhh_f;
    bias[idx] = bi[layer * G4 + n] + bh[layer * G4 + n];
}

__global__ void init_state(const float* __restrict__ enc_h, const float* __restrict__ enc_c,
                           float* __restrict__ hf32, float* __restrict__ cf32,
                           _Float16* __restrict__ h0f, _Float16* __restrict__ h1f) {
    int idx = blockIdx.x * blockDim.x + threadIdx.x;    // < 2*2*B*H = 65536  [dir][j][b][n]
    int n = idx & 511;
    int b = (idx >> 9) & 31;
    int j = (idx >> 14) & 1;
    int dir = idx >> 15;
    size_t eoff = (size_t)j * Bb * 2 * Hh + (size_t)b * 2 * Hh + dir * Hh + n;
    float hv = enc_h[eoff];
    float cv = enc_c[eoff];
    size_t off = (size_t)(dir * 2 + j) * Bb * Hh + (size_t)b * Hh + n;
    hf32[off] = hv;
    cf32[off] = cv;
    // first step reads parity buffer 1
    size_t foff = (size_t)(dir * 2 + 1) * Bb * Hh + (size_t)b * Hh + n;
    if (j == 0) h0f[foff] = (_Float16)hv; else h1f[foff] = (_Float16)hv;
}

// ---------------- fragment set for 2-deep pipeline ----------------

struct Frags {
    h8 al, ah;
    h8 bl0, bh0, bl1, bh1, bl2, bh2, bl3, bh3;
};

__device__ __forceinline__ void load_frags(Frags& f, const _Float16* __restrict__ rowp, int kA0,
                                           const _Float16* __restrict__ bp) {
    f.al  = *(const h8*)(rowp + kA0);
    f.ah  = *(const h8*)(rowp + kA0 + 8);
    f.bl0 = *(const h8*)(bp);
    f.bh0 = *(const h8*)(bp + 8);
    f.bl1 = *(const h8*)(bp + QSTRIDE);
    f.bh1 = *(const h8*)(bp + QSTRIDE + 8);
    f.bl2 = *(const h8*)(bp + 2 * QSTRIDE);
    f.bh2 = *(const h8*)(bp + 2 * QSTRIDE + 8);
    f.bl3 = *(const h8*)(bp + 3 * QSTRIDE);
    f.bh3 = *(const h8*)(bp + 3 * QSTRIDE + 8);
}

#define SHUF16(lo, hi) __builtin_shufflevector(lo, hi, 0,1,2,3,4,5,6,7,8,9,10,11,12,13,14,15)

// ---------------- pipelined step kernel: layer0 @ t=k  +  layer1 @ t=k-1 ----------------

__global__ __launch_bounds__(256) void lstm_step(
    int k,
    const _Float16* __restrict__ xh, const _Float16* __restrict__ wp,
    const float* __restrict__ bias,
    _Float16* __restrict__ h0f, _Float16* __restrict__ h1f,
    float* __restrict__ hf32, float* __restrict__ cf32,
    float* __restrict__ out)
{
    const int w = blockIdx.x * 8 + (threadIdx.x >> 5);  // 0..255 global wave id (wave-uniform)
    const int layer = w >> 7;
    if (layer == 0 && k >= Ss) return;
    if (layer == 1 && k == 0)  return;
    const int t = layer ? (k - 1) : k;
    const int wid   = w & 127;
    const int dir   = wid >> 6;
    const int rem   = wid & 63;
    const int mTile = rem >> 5;
    const int nQuad = rem & 31;

    const int lane = threadIdx.x & 31;
    const int lrow = lane & 15;
    const int hi   = lane >> 4;
    const int m    = mTile * 16 + lrow;
    const int kA0  = hi ? 16 : 0;
    const int tt   = dir ? (Ss - 1 - t) : t;

    const _Float16* aRow0;      // K 0..511 source, row m
    const _Float16* aRow1;      // K 512..1023 source, row m
    _Float16* hOutF16;
    if (layer == 0) {
        aRow0   = xh  + (size_t)m * Ss * Dd + (size_t)tt * Dd;
        aRow1   = h0f + (size_t)(dir * 2 + ((t + 1) & 1)) * Bb * Hh + (size_t)m * Hh;
        hOutF16 = h0f + (size_t)(dir * 2 + (t & 1)) * Bb * Hh;
    } else {
        aRow0   = h0f + (size_t)(dir * 2 + (t & 1)) * Bb * Hh + (size_t)m * Hh;        // h0 new @ t
        aRow1   = h1f + (size_t)(dir * 2 + ((t + 1) & 1)) * Bb * Hh + (size_t)m * Hh;  // h1 old
        hOutF16 = h1f + (size_t)(dir * 2 + (t & 1)) * Bb * Hh;
    }
    const _Float16* wsel  = wp + (size_t)(layer * 2 + dir) * FRAG;
    const _Float16* bBase = wsel + ((size_t)nQuad * KT * 32 + lane) * 16;

    v8f acc[4] = {};
    Frags fA, fB;

    // prologue: fragments for kt = 0
    load_frags(fA, aRow0, kA0, bBase);

#pragma unroll
    for (int kt = 0; kt < KT; ++kt) {
        Frags& cur = (kt & 1) ? fB : fA;
        Frags& nxt = (kt & 1) ? fA : fB;
        if (kt + 1 < KT) {
            const int k2 = kt + 1;   // compile-time under full unroll -> no cndmask
            const _Float16* rp = (k2 < 16) ? (aRow0 + k2 * 32) : (aRow1 + (k2 - 16) * 32);
            load_frags(nxt, rp, kA0, bBase + (size_t)k2 * 512);
        }
        v16h a  = SHUF16(cur.al, cur.ah);
        v16h b0 = SHUF16(cur.bl0, cur.bh0);
        acc[0] = __builtin_amdgcn_wmma_f32_16x16x32_f16(false, a, false, b0, (short)0, acc[0], false, false);
        v16h b1 = SHUF16(cur.bl1, cur.bh1);
        acc[1] = __builtin_amdgcn_wmma_f32_16x16x32_f16(false, a, false, b1, (short)0, acc[1], false, false);
        v16h b2 = SHUF16(cur.bl2, cur.bh2);
        acc[2] = __builtin_amdgcn_wmma_f32_16x16x32_f16(false, a, false, b2, (short)0, acc[2], false, false);
        v16h b3 = SHUF16(cur.bl3, cur.bh3);
        acc[3] = __builtin_amdgcn_wmma_f32_16x16x32_f16(false, a, false, b3, (short)0, acc[3], false, false);
    }

    // fused LSTM elementwise: this wave owns (16 batch x 16 hidden) and all 4 gate tiles
    const int n = nQuad * 16 + lrow;
    const int bbase = (layer * 2 + dir) * G4;
    const float bI = bias[bbase + 0 * Hh + n];
    const float bF = bias[bbase + 1 * Hh + n];
    const float bG = bias[bbase + 2 * Hh + n];
    const float bO = bias[bbase + 3 * Hh + n];
    float* cp = cf32 + (size_t)(dir * 2 + layer) * Bb * Hh;
    float* hp = hf32 + (size_t)(dir * 2 + layer) * Bb * Hh;

#pragma unroll
    for (int r = 0; r < 8; ++r) {
        int b = mTile * 16 + r + (hi << 3);          // C/D layout: M = r + 8*(lane>=16)
        float gi = sigf(acc[0][r] + bI);
        float gf = sigf(acc[1][r] + bF);
        float gg = tanh_(acc[2][r] + bG);
        float go = sigf(acc[3][r] + bO);
        size_t off = (size_t)b * Hh + n;
        float cn = gf * cp[off] + gi * gg;
        float hn = go * tanh_(cn);
        cp[off] = cn;
        hp[off] = hn;
        hOutF16[off] = (_Float16)hn;
        if (layer) out[(size_t)b * (Ss * 2 * Hh) + (size_t)tt * (2 * Hh) + dir * Hh + n] = hn;
    }
}

__global__ void final_copy(const float* __restrict__ hf32, const float* __restrict__ cf32,
                           float* __restrict__ dst) {
    int idx = blockIdx.x * blockDim.x + threadIdx.x;  // < 2*L*B*2H = 131072
    int which = idx >> 16;
    int rem = idx & 65535;
    int j = rem >> 15;
    int b = (rem >> 10) & 31;
    int col = rem & 1023;
    int dir = col >> 9;
    int n = col & 511;
    const float* src = which ? cf32 : hf32;
    dst[idx] = src[(size_t)(dir * 2 + j) * Bb * Hh + (size_t)b * Hh + n];
}

// ---------------- launch ----------------

extern "C" void kernel_launch(void* const* d_in, const int* in_sizes, int n_in,
                              void* d_out, int out_size, void* d_ws, size_t ws_size,
                              hipStream_t stream) {
    const float* x     = (const float*)d_in[0];
    const float* enc_h = (const float*)d_in[1];
    const float* enc_c = (const float*)d_in[2];
    const float* Wih_f = (const float*)d_in[3];
    const float* Whh_f = (const float*)d_in[4];
    const float* bih_f = (const float*)d_in[5];
    const float* bhh_f = (const float*)d_in[6];
    const float* Wih_b = (const float*)d_in[7];
    const float* Whh_b = (const float*)d_in[8];
    const float* bih_b = (const float*)d_in[9];
    const float* bhh_b = (const float*)d_in[10];

    // workspace carve-up (~34.4 MB, all naturally aligned)
    _Float16* xh  = (_Float16*)d_ws;                       // B*S*D halves
    _Float16* wp  = xh + (size_t)Bb * Ss * Dd;             // 4*FRAG halves
    float* bias   = (float*)(wp + 4 * (size_t)FRAG);       // 2*2*4H f32
    _Float16* h0f = (_Float16*)(bias + 4 * G4);            // [dir][parity][B][H]
    _Float16* h1f = h0f + 4 * (size_t)Bb * Hh;
    float* hf32   = (float*)(h1f + 4 * (size_t)Bb * Hh);   // [dir][layer][B][H]
    float* cf32   = hf32 + 4 * (size_t)Bb * Hh;

    float* out = (float*)d_out;

    cvt_x<<<2048, 256, 0, stream>>>(x, xh, Bb * Ss * Dd);
    pack_w<<<(4 * FRAG) / 256, 256, 0, stream>>>(Wih_f, Whh_f, Wih_b, Whh_b, wp);
    pack_bias<<<(4 * G4) / 256, 256, 0, stream>>>(bih_f, bhh_f, bih_b, bhh_b, bias);
    init_state<<<(4 * Bb * Hh) / 256, 256, 0, stream>>>(enc_h, enc_c, hf32, cf32, h0f, h1f);

    // software-pipelined recurrence: kernel k does layer0[t=k] + layer1[t=k-1], both directions
    for (int k = 0; k <= Ss; ++k)
        lstm_step<<<32, 256, 0, stream>>>(k, xh, wp, bias, h0f, h1f, hf32, cf32, out);

    final_copy<<<(2 * Ll * Bb * 2 * Hh) / 256, 256, 0, stream>>>(
        hf32, cf32, out + (size_t)Bb * Ss * 2 * Hh);
}
